// LSTMCell_72035191488637
// MI455X (gfx1250) — compile-verified, matrix operand order checked by reading, not run on hardware
//
#include <hip/hip_runtime.h>
#include <hip/hip_bf16.h>
#include <math.h>

typedef __attribute__((ext_vector_type(16))) _Float16 v16h;
typedef __attribute__((ext_vector_type(8)))  float    v8f;
typedef __attribute__((ext_vector_type(4)))  float    v4f;

union V16U { v16h h; v4f f[2]; };

#define BATCH   32768
#define KDIM    512          // I + H
#define NDIM    1024         // 4*H
#define HDIM    256
#define MTILE   32           // rows per block (2 x 16-row WMMA subtiles)
#define LDS_STRIDE 520       // 512 halfs + 8 pad halfs (1040 B, 16B aligned, bank-balanced)

// ---- quantizers matching quant_clip (jnp.round == round-half-even == rintf) ----
__device__ __forceinline__ float quant_s8(float y) {   // bits=8, sign=True -> scale 128
    y = fminf(fmaxf(y, 0.0f), 1.0f);
    return rintf(y * 128.0f) * (1.0f / 128.0f);
}
__device__ __forceinline__ float quant_u8(float y) {   // bits=8, sign=False -> scale 256
    y = fminf(fmaxf(y, 0.0f), 1.0f);
    return rintf(y * 256.0f) * (1.0f / 256.0f);
}

// ---- hardware transcendentals (v_exp_f32 / v_rcp_f32 / v_tanh_f32) ----
__device__ __forceinline__ float hw_exp(float x) {
    return __builtin_amdgcn_exp2f(x * 1.4426950408889634f);
}
__device__ __forceinline__ float hw_rcp(float x) {
    return __builtin_amdgcn_rcpf(x);
}
__device__ __forceinline__ float hw_sigmoid(float x) {
    return hw_rcp(1.0f + hw_exp(-x));
}
__device__ __forceinline__ float hw_tanh(float x) {
#if __has_builtin(__builtin_amdgcn_tanhf)
    return __builtin_amdgcn_tanhf(x);
#elif __has_builtin(__builtin_amdgcn_tanh_f32)
    return __builtin_amdgcn_tanh_f32(x);
#else
    // tanh(x) = 1 - 2/(exp(2x)+1); saturates correctly: exp->inf => 1, exp->0 => -1
    float e = hw_exp(2.0f * x);
    return 1.0f - 2.0f * hw_rcp(e + 1.0f);
#endif
}

// ---------------- Kernel 1: max(weight_ih), max(weight_hh), pre-scaled by 0.1 ----
__global__ __launch_bounds__(256) void k_max(const float* __restrict__ wih,
                                             const float* __restrict__ whh,
                                             float* __restrict__ scales) {
    __shared__ float red[256];
    const float* src = (blockIdx.x == 0) ? wih : whh;
    float m = -3.4e38f;
    for (int i = threadIdx.x; i < NDIM * HDIM; i += 256)
        m = fmaxf(m, src[i]);
    red[threadIdx.x] = m;
    __syncthreads();
    for (int s = 128; s > 0; s >>= 1) {
        if (threadIdx.x < s) red[threadIdx.x] = fmaxf(red[threadIdx.x], red[threadIdx.x + s]);
        __syncthreads();
    }
    if (threadIdx.x == 0) scales[blockIdx.x] = red[0] * 0.1f;  // NOISE_LEVEL folded in
}

// ---------------- Kernel 2: Wt[n][k] f16 (K-contiguous) + bias_q[n] f32 ---------
__global__ __launch_bounds__(256) void k_prep(const float* __restrict__ wih,
                                              const float* __restrict__ whh,
                                              const float* __restrict__ bih,
                                              const float* __restrict__ bhh,
                                              const float* __restrict__ nih,
                                              const float* __restrict__ nhh,
                                              const float* __restrict__ scales,
                                              _Float16* __restrict__ Wt,
                                              float* __restrict__ biasq) {
    int gidx = blockIdx.x * 256 + threadIdx.x;
    if (gidx < NDIM * KDIM) {
        int n = gidx >> 9;        // 0..1023
        int k = gidx & 511;       // 0..511
        float w, nz, s;
        if (k < HDIM) {           // input half: weight_ih.T[k][n] = weight_ih[n][k]
            w  = wih[n * HDIM + k];
            nz = nih[k * NDIM + n];
            s  = scales[0];
        } else {
            int kk = k - HDIM;
            w  = whh[n * HDIM + kk];
            nz = nhh[kk * NDIM + n];
            s  = scales[1];
        }
        Wt[(size_t)n * KDIM + k] = (_Float16)(quant_s8(w) + nz * s);
    }
    if (gidx < NDIM)
        biasq[gidx] = quant_s8(bih[gidx]) + quant_s8(bhh[gidx]);
}

// ---------------- Kernel 3: fused GEMM (WMMA f16->f32) + LSTM epilogue ----------
__global__ __launch_bounds__(256) void k_main(const float* __restrict__ input,
                                              const float* __restrict__ hx,
                                              const float* __restrict__ cx,
                                              const _Float16* __restrict__ Wt,
                                              const float* __restrict__ biasq,
                                              float* __restrict__ out) {
    __shared__ _Float16 As[MTILE * LDS_STRIDE];
    const int tid = threadIdx.x;
    const int m0  = blockIdx.x * MTILE;

    // ---- Stage A tile [32 rows x 512 K] = [input | hx] as f16 into LDS ----
    {
        int r  = tid >> 3;            // 0..31
        int c0 = (tid & 7) * 64;      // 0,64,...,448 (64-chunks never cross the 256 split)
        const float* src = (c0 < HDIM)
            ? (input + (size_t)(m0 + r) * HDIM + c0)
            : (hx    + (size_t)(m0 + r) * HDIM + (c0 - HDIM));
        _Float16* dst = &As[r * LDS_STRIDE + c0];
        #pragma unroll
        for (int i = 0; i < 64; i += 4) {
            v4f v = *reinterpret_cast<const v4f*>(src + i);
            dst[i + 0] = (_Float16)v[0];
            dst[i + 1] = (_Float16)v[1];
            dst[i + 2] = (_Float16)v[2];
            dst[i + 3] = (_Float16)v[3];
        }
    }
    __syncthreads();

    const int wave = tid >> 5;       // 0..7
    const int lane = tid & 31;
    const int lhi  = lane >> 4;      // 0/1 (K-half / M-half selector)
    const int llo  = lane & 15;      // matrix row (A) / column (B,C,D)

    // Wave w owns h-tiles {2w, 2w+1}; for each, all four gates: n = g*256 + htile*16
    int nbase[8];
    #pragma unroll
    for (int g = 0; g < 4; ++g)
        #pragma unroll
        for (int t = 0; t < 2; ++t)
            nbase[g * 2 + t] = g * HDIM + (wave * 2 + t) * 16;

    // Warm the L2 path for this wave's Wt columns
    #pragma unroll
    for (int i = 0; i < 8; ++i)
        __builtin_prefetch(Wt + (size_t)(nbase[i] + llo) * KDIM, 0, 1);

    // acc[i*2 + s]: output tile i (gate/h-tile), m-subtile s (rows m0+16s .. +16)
    v8f acc[16];
    #pragma unroll
    for (int i = 0; i < 16; ++i)
        #pragma unroll
        for (int j = 0; j < 8; ++j) acc[i][j] = 0.0f;

    // ---- K loop: 16 WMMA steps; each B-fragment feeds 2 WMMAs (m-subtiles) ----
    for (int k0 = 0; k0 < KDIM; k0 += 32) {
        V16U a0, a1;
        const _Float16* ar0 = &As[llo * LDS_STRIDE + k0 + lhi * 8];
        const _Float16* ar1 = ar0 + 16 * LDS_STRIDE;
        a0.f[0] = *reinterpret_cast<const v4f*>(ar0);
        a0.f[1] = *reinterpret_cast<const v4f*>(ar0 + 16);
        a1.f[0] = *reinterpret_cast<const v4f*>(ar1);
        a1.f[1] = *reinterpret_cast<const v4f*>(ar1 + 16);
        #pragma unroll
        for (int i = 0; i < 8; ++i) {
            V16U b;
            const _Float16* bp = Wt + (size_t)(nbase[i] + llo) * KDIM + k0 + lhi * 8;
            b.f[0] = *reinterpret_cast<const v4f*>(bp);
            b.f[1] = *reinterpret_cast<const v4f*>(bp + 16);
            acc[i * 2 + 0] = __builtin_amdgcn_wmma_f32_16x16x32_f16(
                false, a0.h, false, b.h, (short)0, acc[i * 2 + 0], false, false);
            acc[i * 2 + 1] = __builtin_amdgcn_wmma_f32_16x16x32_f16(
                false, a1.h, false, b.h, (short)0, acc[i * 2 + 1], false, false);
        }
    }

    // ---- Epilogue: bias + activations + quant + cy/hy, all in-register ----
    const size_t HY_ELEMS = (size_t)BATCH * HDIM;
    #pragma unroll
    for (int t = 0; t < 2; ++t) {
        int h = (wave * 2 + t) * 16 + llo;   // hidden index 0..255
        float bi = biasq[0 * HDIM + h];
        float bf = biasq[1 * HDIM + h];
        float bc = biasq[2 * HDIM + h];
        float bo = biasq[3 * HDIM + h];
        #pragma unroll
        for (int s = 0; s < 2; ++s) {
            v8f ai = acc[(0 * 2 + t) * 2 + s];
            v8f af = acc[(1 * 2 + t) * 2 + s];
            v8f ac = acc[(2 * 2 + t) * 2 + s];
            v8f ao = acc[(3 * 2 + t) * 2 + s];
            #pragma unroll
            for (int j = 0; j < 8; ++j) {
                int row = m0 + s * 16 + j + lhi * 8;  // C/D: vgpr j -> M = j + 8*(lane>=16)
                float gi = quant_u8(hw_sigmoid(ai[j] + bi));
                float gf = quant_u8(hw_sigmoid(af[j] + bf));
                float gc = quant_s8(hw_tanh(ac[j] + bc));
                float go = quant_u8(hw_sigmoid(ao[j] + bo));
                float cv = cx[(size_t)row * HDIM + h];
                float cy = quant_s8(gf * cv + gi * gc);
                float hy = quant_s8(go * hw_tanh(cy));
                out[(size_t)row * HDIM + h]            = hy;
                out[HY_ELEMS + (size_t)row * HDIM + h] = cy;
            }
        }
    }
}

extern "C" void kernel_launch(void* const* d_in, const int* in_sizes, int n_in,
                              void* d_out, int out_size, void* d_ws, size_t ws_size,
                              hipStream_t stream) {
    (void)in_sizes; (void)n_in; (void)out_size; (void)ws_size;
    const float* input = (const float*)d_in[0];
    const float* hx    = (const float*)d_in[1];
    const float* cx    = (const float*)d_in[2];
    const float* wih   = (const float*)d_in[3];
    const float* whh   = (const float*)d_in[4];
    const float* bih   = (const float*)d_in[5];
    const float* bhh   = (const float*)d_in[6];
    const float* nih   = (const float*)d_in[7];
    const float* nhh   = (const float*)d_in[8];

    // Workspace layout (all 16B-aligned offsets):
    //   [0, 1 MB)        : Wt   f16 [1024][512]
    //   [1 MB, +4 KB)    : biasq f32 [1024]
    //   [1 MB+4 KB, +8B) : scales f32 [2]
    _Float16* Wt     = (_Float16*)d_ws;
    float*    biasq  = (float*)((char*)d_ws + (size_t)NDIM * KDIM * 2);
    float*    scales = (float*)((char*)d_ws + (size_t)NDIM * KDIM * 2 + NDIM * 4);

    k_max <<<2, 256, 0, stream>>>(wih, whh, scales);
    k_prep<<<(NDIM * KDIM + 255) / 256, 256, 0, stream>>>(wih, whh, bih, bhh,
                                                          nih, nhh, scales, Wt, biasq);
    k_main<<<BATCH / MTILE, 256, 0, stream>>>(input, hx, cx, Wt, biasq, (float*)d_out);
}
